// PointTransformer_86947317940513
// MI455X (gfx1250) — matrix operand order repaired and checked by reading, not compiled, not run on hardware
//
#include <hip/hip_runtime.h>
#include <hip/hip_bf16.h>

#define N_PTS 65536
#define INC   256
#define MID   64
#define KNB   16
#define BN_EPS 1e-5f

typedef __attribute__((ext_vector_type(16))) _Float16 v16h;
typedef __attribute__((ext_vector_type(8)))  _Float16 v8h;
typedef __attribute__((ext_vector_type(8)))  float    v8f;

static __device__ inline v8f wmma_f16(v16h a, v16h b, v8f c) {
  // D = A(16x32 f16) x B(32x16 f16) + C(16x16 f32)
  return __builtin_amdgcn_wmma_f32_16x16x32_f16(false, a, false, b, (short)0, c,
                                                false, false);
}

// A fragment from row-major f16 [16 x ld], K-chunk of 32 at kbase.
// lane L (g=L/16): row m=L%16, halves K = kbase+g*8..+7 and kbase+16+g*8..+7
static __device__ inline v16h load_a_f16(const _Float16* src, int ld, int lane, int kbase) {
  const int m = lane & 15, g = lane >> 4;
  const _Float16* p0 = src + m * ld + kbase + g * 8;
  v8h lo = *(const v8h*)p0;
  v8h hi = *(const v8h*)(p0 + 16);
  return __builtin_shufflevector(lo, hi, 0,1,2,3,4,5,6,7,8,9,10,11,12,13,14,15);
}

// Same, but source is f32 (converted on the fly).
static __device__ inline v16h load_a_f32(const float* src, int ld, int lane, int kbase) {
  const int m = lane & 15, g = lane >> 4;
  const float* p0 = src + m * ld + kbase + g * 8;
  v8f flo = *(const v8f*)p0;
  v8f fhi = *(const v8f*)(p0 + 16);
  v8h lo = __builtin_convertvector(flo, v8h);
  v8h hi = __builtin_convertvector(fhi, v8h);
  return __builtin_shufflevector(lo, hi, 0,1,2,3,4,5,6,7,8,9,10,11,12,13,14,15);
}

// B fragment: x @ W.T -> B column n == row n of W (contiguous, stride ld).
// lane L (g=L/16): col n=L%16, halves K = kbase+g*8..+7 and kbase+16+g*8..+7
static __device__ inline v16h load_b_f16(const _Float16* w, int ld, int lane, int kbase) {
  const int n = lane & 15, g = lane >> 4;
  const _Float16* p0 = w + n * ld + kbase + g * 8;
  v8h lo = *(const v8h*)p0;
  v8h hi = *(const v8h*)(p0 + 16);
  return __builtin_shufflevector(lo, hi, 0,1,2,3,4,5,6,7,8,9,10,11,12,13,14,15);
}

// exchange with lane^16 (SWAPX16) — wave32-safe cross-half swap
static __device__ inline float swap16(float x) {
  int r = __builtin_amdgcn_ds_swizzle(__builtin_bit_cast(int, x), 0x401F);
  return __builtin_bit_cast(float, r);
}

// ---------------------------------------------------------------------------
// Prep: fold BN into linears, convert weights to f16.
// ---------------------------------------------------------------------------
__global__ void k_prep(
    const float* __restrict__ w_in, const float* __restrict__ w_q,
    const float* __restrict__ w_k, const float* __restrict__ w_v,
    const float* __restrict__ ga_w1, const float* __restrict__ ga_b1,
    const float* __restrict__ ga_g1, const float* __restrict__ ga_be1,
    const float* __restrict__ ga_m1, const float* __restrict__ ga_v1,
    const float* __restrict__ ga_w2, const float* __restrict__ ga_b2,
    const float* __restrict__ ga_g2, const float* __restrict__ ga_be2,
    const float* __restrict__ ga_m2, const float* __restrict__ ga_v2,
    const float* __restrict__ pe_w1, const float* __restrict__ pe_b1,
    const float* __restrict__ pe_g1, const float* __restrict__ pe_be1,
    const float* __restrict__ pe_m1, const float* __restrict__ pe_v1,
    const float* __restrict__ pe_w2, const float* __restrict__ pe_b2,
    const float* __restrict__ pe_g2, const float* __restrict__ pe_be2,
    const float* __restrict__ pe_m2, const float* __restrict__ pe_v2,
    const float* __restrict__ w_out,
    _Float16* __restrict__ w_in_h, _Float16* __restrict__ wq_h,
    _Float16* __restrict__ wk_h, _Float16* __restrict__ wv_h,
    _Float16* __restrict__ ga1_h, _Float16* __restrict__ ga2_h,
    _Float16* __restrict__ w_out_h,
    float* __restrict__ pe1, float* __restrict__ pe_w2f,
    float* __restrict__ pe_b2f, float* __restrict__ ga_b1f,
    float* __restrict__ ga_b2f)
{
  const int t = threadIdx.x;
  for (int i = t; i < MID * INC; i += 256) {
    w_in_h[i]  = (_Float16)w_in[i];
    w_out_h[i] = (_Float16)w_out[i];
  }
  for (int i = t; i < MID * MID; i += 256) {
    wq_h[i] = (_Float16)w_q[i];
    wk_h[i] = (_Float16)w_k[i];
    wv_h[i] = (_Float16)w_v[i];
    const int c = i >> 6;  // output channel (row of W)
    const float s1 = ga_g1[c] * rsqrtf(ga_v1[c] + BN_EPS);
    const float s2 = ga_g2[c] * rsqrtf(ga_v2[c] + BN_EPS);
    ga1_h[i] = (_Float16)(ga_w1[i] * s1);
    ga2_h[i] = (_Float16)(ga_w2[i] * s2);
  }
  if (t < MID) {
    const float s1 = ga_g1[t] * rsqrtf(ga_v1[t] + BN_EPS);
    ga_b1f[t] = ga_b1[t] * s1 + (ga_be1[t] - ga_m1[t] * s1);
    const float s2 = ga_g2[t] * rsqrtf(ga_v2[t] + BN_EPS);
    ga_b2f[t] = ga_b2[t] * s2 + (ga_be2[t] - ga_m2[t] * s2);
    const float sp = pe_g2[t] * rsqrtf(pe_v2[t] + BN_EPS);
    pe_b2f[t] = pe_b2[t] * sp + (pe_be2[t] - pe_m2[t] * sp);
    pe_w2f[t * 4 + 0] = pe_w2[t * 3 + 0] * sp;
    pe_w2f[t * 4 + 1] = pe_w2[t * 3 + 1] * sp;
    pe_w2f[t * 4 + 2] = pe_w2[t * 3 + 2] * sp;
    pe_w2f[t * 4 + 3] = 0.f;
  }
  if (t < 3) {
    const float s = pe_g1[t] * rsqrtf(pe_v1[t] + BN_EPS);
    pe1[t * 3 + 0] = pe_w1[t * 3 + 0] * s;
    pe1[t * 3 + 1] = pe_w1[t * 3 + 1] * s;
    pe1[t * 3 + 2] = pe_w1[t * 3 + 2] * s;
    pe1[9 + t]     = pe_b1[t] * s + (pe_be1[t] - pe_m1[t] * s);
  }
}

// ---------------------------------------------------------------------------
// xx = feats @ w_in.T ; q/k/v = xx @ W.T + b   (one wave = 16 rows)
// ---------------------------------------------------------------------------
__global__ __launch_bounds__(256) void k_qkv(
    const float* __restrict__ feats, const _Float16* __restrict__ w_in_h,
    const _Float16* __restrict__ wq_h, const _Float16* __restrict__ wk_h,
    const _Float16* __restrict__ wv_h,
    const float* __restrict__ b_q, const float* __restrict__ b_k,
    const float* __restrict__ b_v,
    _Float16* __restrict__ q_h, _Float16* __restrict__ k_h,
    _Float16* __restrict__ v_h)
{
  __shared__ __align__(32) _Float16 xxT[8][16 * MID];
  const int wave = threadIdx.x >> 5, lane = threadIdx.x & 31;
  const int rowBase = (blockIdx.x * 8 + wave) * 16;
  const float* fA = feats + (size_t)rowBase * INC;
  __builtin_prefetch(fA + (lane & 15) * INC, 0, 0);

  v8f acc[4] = {};
#pragma unroll
  for (int kc = 0; kc < INC; kc += 32) {
    v16h a = load_a_f32(fA, INC, lane, kc);
#pragma unroll
    for (int ct = 0; ct < 4; ct++)
      acc[ct] = wmma_f16(a, load_b_f16(w_in_h + ct * 16 * INC, INC, lane, kc), acc[ct]);
  }

  _Float16* xs = xxT[wave];
  const int colL = lane & 15, g = lane >> 4;
#pragma unroll
  for (int ct = 0; ct < 4; ct++)
#pragma unroll
    for (int j = 0; j < 8; j++)
      xs[(j + 8 * g) * MID + ct * 16 + colL] = (_Float16)acc[ct][j];

  const _Float16* Ws[3] = {wq_h, wk_h, wv_h};
  const float*    Bs[3] = {b_q, b_k, b_v};
  _Float16*       Ds[3] = {q_h, k_h, v_h};
#pragma unroll
  for (int mtx = 0; mtx < 3; mtx++) {
    v8f c2[4] = {};
#pragma unroll
    for (int kc = 0; kc < MID; kc += 32) {
      v16h a = load_a_f16(xs, MID, lane, kc);
#pragma unroll
      for (int ct = 0; ct < 4; ct++)
        c2[ct] = wmma_f16(a, load_b_f16(Ws[mtx] + ct * 16 * MID, MID, lane, kc), c2[ct]);
    }
#pragma unroll
    for (int ct = 0; ct < 4; ct++) {
      const float bb = Bs[mtx][ct * 16 + colL];
#pragma unroll
      for (int j = 0; j < 8; j++)
        Ds[mtx][(size_t)(rowBase + j + 8 * g) * MID + ct * 16 + colL] =
            (_Float16)(c2[ct][j] + bb);
    }
  }
}

// ---------------------------------------------------------------------------
// Fused: gather + pos-enc + gamma MLP (WMMA) + softmax over K + weighted sum.
// One wave = one point; its 16 neighbors form the 16-row WMMA A tile.
// ---------------------------------------------------------------------------
__global__ __launch_bounds__(256) void k_attn(
    const float* __restrict__ pos, const int* __restrict__ idx,
    const _Float16* __restrict__ q_h, const _Float16* __restrict__ k_h,
    const _Float16* __restrict__ v_h,
    const float* __restrict__ pe1,     // 12 floats: folded 3x3 W + bias
    const float* __restrict__ pe_w2f,  // [64*4] folded
    const float* __restrict__ pe_b2f,  // [64]
    const _Float16* __restrict__ ga1_h, const float* __restrict__ ga_b1f,
    const _Float16* __restrict__ ga2_h, const float* __restrict__ ga_b2f,
    _Float16* __restrict__ mid_h)
{
  __shared__ __align__(32) _Float16 aT[8][16 * MID];
  __shared__ __align__(32) _Float16 xvpT[8][16 * MID];
  __shared__ float s_pw2[MID * 4];
  __shared__ float s_pb2[MID];
  const int wave = threadIdx.x >> 5, lane = threadIdx.x & 31;
  const int n = blockIdx.x * 8 + wave;

  s_pw2[threadIdx.x] = pe_w2f[threadIdx.x];
  if (threadIdx.x < MID) s_pb2[threadIdx.x] = pe_b2f[threadIdx.x];
  __syncthreads();

  _Float16* A   = aT[wave];
  _Float16* XVP = xvpT[wave];

  // Gather + pos_enc: lane handles row r = lane/2, 32 cols at cb.
  const int r = lane >> 1, cb = (lane & 1) * 32;
  const int nid = idx[(size_t)n * KNB + r];
  const float px = pos[nid * 3 + 0], py = pos[nid * 3 + 1], pz = pos[nid * 3 + 2];
  const float h0 = fmaxf(0.f, px * pe1[0] + py * pe1[1] + pz * pe1[2] + pe1[9]);
  const float h1 = fmaxf(0.f, px * pe1[3] + py * pe1[4] + pz * pe1[5] + pe1[10]);
  const float h2 = fmaxf(0.f, px * pe1[6] + py * pe1[7] + pz * pe1[8] + pe1[11]);
  const _Float16* krow = k_h + (size_t)nid * MID + cb;
  const _Float16* vrow = v_h + (size_t)nid * MID + cb;
  const _Float16* qrow = q_h + (size_t)n * MID + cb;
#pragma unroll
  for (int c = 0; c < 32; c++) {
    const int cc = cb + c;
    const float p = fmaxf(0.f, h0 * s_pw2[cc * 4 + 0] + h1 * s_pw2[cc * 4 + 1] +
                               h2 * s_pw2[cc * 4 + 2] + s_pb2[cc]);
    A[r * MID + cc]   = (_Float16)((float)krow[c] - (float)qrow[c] + p);
    XVP[r * MID + cc] = (_Float16)((float)vrow[c] + p);
  }

  // gamma MLP layer 1: 16x64 @ 64x64
  v8f c1[4] = {};
#pragma unroll
  for (int kc = 0; kc < MID; kc += 32) {
    v16h a = load_a_f16(A, MID, lane, kc);
#pragma unroll
    for (int ct = 0; ct < 4; ct++)
      c1[ct] = wmma_f16(a, load_b_f16(ga1_h + ct * 16 * MID, MID, lane, kc), c1[ct]);
  }
  const int colL = lane & 15, g = lane >> 4;
#pragma unroll
  for (int ct = 0; ct < 4; ct++) {
    const float bb = ga_b1f[ct * 16 + colL];
#pragma unroll
    for (int j = 0; j < 8; j++)
      A[(j + 8 * g) * MID + ct * 16 + colL] = (_Float16)fmaxf(0.f, c1[ct][j] + bb);
  }

  // gamma MLP layer 2
  v8f c2[4] = {};
#pragma unroll
  for (int kc = 0; kc < MID; kc += 32) {
    v16h a = load_a_f16(A, MID, lane, kc);
#pragma unroll
    for (int ct = 0; ct < 4; ct++)
      c2[ct] = wmma_f16(a, load_b_f16(ga2_h + ct * 16 * MID, MID, lane, kc), c2[ct]);
  }

  // relu+bias -> softmax over the 16 neighbor rows -> weighted sum of (xv+p)
#pragma unroll
  for (int ct = 0; ct < 4; ct++) {
    const float bb = ga_b2f[ct * 16 + colL];
    float x[8];
#pragma unroll
    for (int j = 0; j < 8; j++) x[j] = fmaxf(0.f, c2[ct][j] + bb);
    float m = x[0];
#pragma unroll
    for (int j = 1; j < 8; j++) m = fmaxf(m, x[j]);
    m = fmaxf(m, swap16(m));                     // both half-tiles of the column
    float e[8], ssum = 0.f;
#pragma unroll
    for (int j = 0; j < 8; j++) { e[j] = __expf(x[j] - m); ssum += e[j]; }
    ssum += swap16(ssum);
    const float inv = 1.0f / ssum;
    float part = 0.f;
#pragma unroll
    for (int j = 0; j < 8; j++)
      part += e[j] * (float)XVP[(j + 8 * g) * MID + ct * 16 + colL];
    part *= inv;
    part += swap16(part);
    if (g == 0) mid_h[(size_t)n * MID + ct * 16 + colL] = (_Float16)part;
  }
}

// ---------------------------------------------------------------------------
// out = mid @ w_out.T + feats   (one wave = 16 rows x 256 cols)
// ---------------------------------------------------------------------------
__global__ __launch_bounds__(256) void k_out(
    const _Float16* __restrict__ mid_h, const _Float16* __restrict__ w_out_h,
    const float* __restrict__ feats, float* __restrict__ out)
{
  __shared__ __align__(32) float st[8][16 * 16];
  const int wave = threadIdx.x >> 5, lane = threadIdx.x & 31;
  const int rowBase = (blockIdx.x * 8 + wave) * 16;
  const int colL = lane & 15, g = lane >> 4;
  const _Float16* Am = mid_h + (size_t)rowBase * MID;
  __builtin_prefetch(feats + (size_t)(rowBase + (lane & 15)) * INC, 0, 0);
  v16h a0 = load_a_f16(Am, MID, lane, 0);
  v16h a1 = load_a_f16(Am, MID, lane, 32);
  float* s = st[wave];
  const int r2 = lane >> 1, cb2 = (lane & 1) * 8;
#pragma unroll
  for (int ct = 0; ct < 16; ct++) {
    v8f c = {};
    c = wmma_f16(a0, load_b_f16(w_out_h + ct * 16 * MID, MID, lane, 0), c);
    c = wmma_f16(a1, load_b_f16(w_out_h + ct * 16 * MID, MID, lane, 32), c);
#pragma unroll
    for (int j = 0; j < 8; j++) s[(j + 8 * g) * 16 + colL] = c[j];
    // coalesced writeout with residual add (same-wave LDS ordering is in-order)
    const float* fp = feats + (size_t)(rowBase + r2) * INC + ct * 16 + cb2;
    float*       op = out   + (size_t)(rowBase + r2) * INC + ct * 16 + cb2;
#pragma unroll
    for (int t = 0; t < 8; t++) op[t] = s[r2 * 16 + cb2 + t] + fp[t];
  }
}

// ---------------------------------------------------------------------------
extern "C" void kernel_launch(void* const* d_in, const int* in_sizes, int n_in,
                              void* d_out, int out_size, void* d_ws, size_t ws_size,
                              hipStream_t stream) {
  const float* feats = (const float*)d_in[0];
  const float* pos   = (const float*)d_in[1];
  const int*   idx   = (const int*)d_in[2];
  const float* w_in  = (const float*)d_in[3];
  const float* w_q = (const float*)d_in[4];  const float* b_q = (const float*)d_in[5];
  const float* w_k = (const float*)d_in[6];  const float* b_k = (const float*)d_in[7];
  const float* w_v = (const float*)d_in[8];  const float* b_v = (const float*)d_in[9];
  const float* pe_w1 = (const float*)d_in[10]; const float* pe_b1 = (const float*)d_in[11];
  const float* pe_g1 = (const float*)d_in[12]; const float* pe_be1 = (const float*)d_in[13];
  const float* pe_m1 = (const float*)d_in[14]; const float* pe_v1 = (const float*)d_in[15];
  const float* pe_w2 = (const float*)d_in[16]; const float* pe_b2 = (const float*)d_in[17];
  const float* pe_g2 = (const float*)d_in[18]; const float* pe_be2 = (const float*)d_in[19];
  const float* pe_m2 = (const float*)d_in[20]; const float* pe_v2 = (const float*)d_in[21];
  const float* ga_w1 = (const float*)d_in[22]; const float* ga_b1 = (const float*)d_in[23];
  const float* ga_g1 = (const float*)d_in[24]; const float* ga_be1 = (const float*)d_in[25];
  const float* ga_m1 = (const float*)d_in[26]; const float* ga_v1 = (const float*)d_in[27];
  const float* ga_w2 = (const float*)d_in[28]; const float* ga_b2 = (const float*)d_in[29];
  const float* ga_g2 = (const float*)d_in[30]; const float* ga_be2 = (const float*)d_in[31];
  const float* ga_m2 = (const float*)d_in[32]; const float* ga_v2 = (const float*)d_in[33];
  const float* w_out = (const float*)d_in[34];
  float* out = (float*)d_out;

  char* base = (char*)d_ws;
  size_t off = 0;
  auto take = [&](size_t bytes) -> void* {
    void* p = base + off;
    off += (bytes + 255) & ~(size_t)255;
    return p;
  };
  _Float16* q_h   = (_Float16*)take((size_t)N_PTS * MID * 2);
  _Float16* k_h   = (_Float16*)take((size_t)N_PTS * MID * 2);
  _Float16* v_h   = (_Float16*)take((size_t)N_PTS * MID * 2);
  _Float16* mid_h = (_Float16*)take((size_t)N_PTS * MID * 2);
  _Float16* w_in_h  = (_Float16*)take(MID * INC * 2);
  _Float16* wq_h    = (_Float16*)take(MID * MID * 2);
  _Float16* wk_h    = (_Float16*)take(MID * MID * 2);
  _Float16* wv_h    = (_Float16*)take(MID * MID * 2);
  _Float16* ga1_h   = (_Float16*)take(MID * MID * 2);
  _Float16* ga2_h   = (_Float16*)take(MID * MID * 2);
  _Float16* w_out_h = (_Float16*)take(INC * MID * 2);
  float* pe1    = (float*)take(12 * 4);
  float* pe_w2f = (float*)take(MID * 4 * 4);
  float* pe_b2f = (float*)take(MID * 4);
  float* ga_b1f = (float*)take(MID * 4);
  float* ga_b2f = (float*)take(MID * 4);

  k_prep<<<1, 256, 0, stream>>>(
      w_in, w_q, w_k, w_v,
      ga_w1, ga_b1, ga_g1, ga_be1, ga_m1, ga_v1,
      ga_w2, ga_b2, ga_g2, ga_be2, ga_m2, ga_v2,
      pe_w1, pe_b1, pe_g1, pe_be1, pe_m1, pe_v1,
      pe_w2, pe_b2, pe_g2, pe_be2, pe_m2, pe_v2,
      w_out,
      w_in_h, wq_h, wk_h, wv_h, ga1_h, ga2_h, w_out_h,
      pe1, pe_w2f, pe_b2f, ga_b1f, ga_b2f);

  k_qkv<<<N_PTS / 128, 256, 0, stream>>>(feats, w_in_h, wq_h, wk_h, wv_h,
                                         b_q, b_k, b_v, q_h, k_h, v_h);

  k_attn<<<N_PTS / 8, 256, 0, stream>>>(pos, idx, q_h, k_h, v_h,
                                        pe1, pe_w2f, pe_b2f,
                                        ga1_h, ga_b1f, ga2_h, ga_b2f, mid_h);

  k_out<<<N_PTS / 128, 256, 0, stream>>>(mid_h, w_out_h, feats, out);

  (void)in_sizes; (void)n_in; (void)out_size; (void)ws_size;
}
